// FeatureLossOursBMSE_37477884625163
// MI455X (gfx1250) — matrix operand order repaired and checked by reading, not compiled, not run on hardware
//
#include <hip/hip_runtime.h>
#include <hip/hip_bf16.h>
#include <math.h>

// ---------------------------------------------------------------------------
// N=8, Cs=128, Ct=256, H=W=64.
// Pipeline (all on one stream):
//   prep:  zero halo tensors; preds_S -> NHWC bf16; weights -> k-major bf16
//   conv0: 1x1  (K=128)  -> masked  NHWC-padded bf16 [8][66][66][256], *checker
//   conv1: 3x3  (K=2304) -> t1      NHWC-padded bf16, relu
//   conv2: 3x3  (K=2304) -> new_fea NCHW f32
//   loss:  deterministic 8x8 Gram + norms + logsumexp -> scalar
// Conv = implicit GEMM, v_wmma_f32_16x16x32_bf16, async global->LDS staging,
// double-buffered LDS, 2x2 WMMA tiles per wave (32co x 32px).
// ---------------------------------------------------------------------------

typedef __attribute__((ext_vector_type(16))) __bf16 v16bf;
typedef __attribute__((ext_vector_type(8)))  float  v8f;
typedef __attribute__((ext_vector_type(4)))  int    v4i_;

#if __has_builtin(__builtin_amdgcn_global_load_async_to_lds_b128)
#define HAVE_ASYNC_LDS 1
typedef __attribute__((address_space(1))) v4i_ as1_v4i;   // global (prints __device__)
typedef __attribute__((address_space(3))) v4i_ as3_v4i;   // LDS    (prints __shared__)
#endif

__device__ __forceinline__ unsigned short f2bf(float f) {
  unsigned int u = __float_as_uint(f);
  u += 0x7FFFu + ((u >> 16) & 1u);           // RNE f32 -> bf16
  return (unsigned short)(u >> 16);
}

__device__ __forceinline__ void async_copy16(const unsigned short* g, unsigned short* l) {
#if defined(HAVE_ASYNC_LDS)
  __builtin_amdgcn_global_load_async_to_lds_b128(
      (as1_v4i*)(void*)g, (as3_v4i*)(void*)l, 0, 0);
#else
  *(uint4*)l = *(const uint4*)g;
#endif
}

template<int N>
__device__ __forceinline__ void wait_async() {
#if __has_builtin(__builtin_amdgcn_s_wait_asynccnt)
  __builtin_amdgcn_s_wait_asynccnt(N);
#else
  asm volatile("s_wait_asynccnt %0" :: "i"(N) : "memory");
#endif
}

// ---- LDS fragment gathers per documented CDNA5 bf16 WMMA layouts -----------
// rows padded to 40 shorts (80B) to spread LDS banks; all reads ds_load_b128.
__device__ __forceinline__ v16bf ldA(const unsigned short* base, int row, int lh) {
  union { v16bf v; uint4 q[2]; } u;
  const char* p = (const char*)(base + row * 40);
  u.q[0] = *(const uint4*)(p + lh * 16);        // K = lh*8 + 0..7
  u.q[1] = *(const uint4*)(p + 32 + lh * 16);   // K = 16 + lh*8 + 0..7
  return u.v;
}
__device__ __forceinline__ v16bf ldB(const unsigned short* base, int row, int lh) {
  union { v16bf v; uint4 q[2]; } u;
  const char* p = (const char*)(base + row * 40);
  u.q[0] = *(const uint4*)(p + lh * 16);        // K = lh*8 + 0..7
  u.q[1] = *(const uint4*)(p + 32 + lh * 16);   // K = 16 + lh*8 + 0..7
  return u.v;
}

// ---------------------------------------------------------------------------
// Staging: one 32-wide K chunk. A: 128co x 64B (2 b128/lane). B: 64px x 64B
// (1 b128/lane). NHWC + halo => contiguous, no bounds checks, no converts.
// ---------------------------------------------------------------------------
template<int LAYER>
__device__ __forceinline__ void stage(const unsigned short* __restrict__ Xb,
                                      const unsigned short* __restrict__ Wr,
                                      unsigned short* Ab, unsigned short* Bb,
                                      int tid, int co0, int img, int h, int kk)
{
  constexpr int KW = (LAYER == 0) ? 128 : 2304;
  const int tap = (LAYER == 0) ? 0 : (kk >> 8);
  const int ci0 = (LAYER == 0) ? kk : (kk & 255);
  const int dy = tap / 3 - 1, dx = tap % 3 - 1;

#pragma unroll
  for (int t = 0; t < 2; ++t) {                 // A tile: weights (k-major)
    int task = tid + t * 256;
    int co = task >> 2, seg = task & 3;
    const unsigned short* g = Wr + (size_t)(co0 + co) * KW + kk + seg * 8;
    async_copy16(g, Ab + co * 40 + seg * 8);
  }
  {                                             // B tile: activations (NHWC)
    int px = tid >> 2, seg = tid & 3;
    size_t pix;
    if (LAYER == 0) pix = ((size_t)img * 4096 + h * 64 + px) * 128 + ci0;
    else            pix = (((size_t)img * 66 + (h + dy + 1)) * 66 + (px + dx + 1)) * 256 + ci0;
    async_copy16(Xb + pix + seg * 8, Bb + px * 40 + seg * 8);
  }
}

// ---------------------------------------------------------------------------
// Conv kernel. Block 256 = 8 waves; block tile 128co x 64px (one image row);
// wave tile 32co x 32px = 2x2 WMMA tiles. Grid (8*64 rows, 256/128 co-tiles).
// ---------------------------------------------------------------------------
template<int LAYER>
__global__ __launch_bounds__(256, 2)
void conv_wmma_kernel(const unsigned short* __restrict__ Xb,
                      const unsigned short* __restrict__ Wr,
                      const float* __restrict__ Bias,
                      void* __restrict__ Y)
{
  constexpr int KW = (LAYER == 0) ? 128 : 2304;

  __shared__ __align__(16) unsigned short Ash[2][128 * 40];
  __shared__ __align__(16) unsigned short Bsh[2][64 * 40];

  const int tid = threadIdx.x;
  const int img = blockIdx.x >> 6;
  const int h   = blockIdx.x & 63;
  const int co0 = blockIdx.y * 128;

  const int lane = tid & 31;
  const int wid  = tid >> 5;
  const int lh   = lane >> 4;
  const int lr   = lane & 15;
  const int mw2  = (wid >> 1) * 32;   // co sub-tile base within 128
  const int nw2  = (wid & 1) * 32;    // px sub-tile base within 64

  v8f c00 = {}, c01 = {}, c10 = {}, c11 = {};

  stage<LAYER>(Xb, Wr, Ash[0], Bsh[0], tid, co0, img, h, 0);
  int buf = 0;
  for (int kk = 0; kk < KW; kk += 32) {
    const int nxt = buf ^ 1;
    const bool more = (kk + 32 < KW);
    if (more) {
      stage<LAYER>(Xb, Wr, Ash[nxt], Bsh[nxt], tid, co0, img, h, kk + 32);
      if (kk + 64 < KW)   // pull K+2 weight chunk toward L2 (global_prefetch)
        __builtin_prefetch(Wr + (size_t)(co0 + (tid >> 1)) * KW + kk + 64, 0, 0);
      wait_async<3>();    // current chunk's 3 async ops done; next 3 in flight
    } else {
      wait_async<0>();
    }
    __syncthreads();

    const unsigned short* Ab = Ash[buf];
    const unsigned short* Bb = Bsh[buf];
    v16bf a0 = ldA(Ab, mw2 + lr,      lh);
    v16bf a1 = ldA(Ab, mw2 + 16 + lr, lh);
    v16bf b0 = ldB(Bb, nw2 + lr,      lh);
    v16bf b1 = ldB(Bb, nw2 + 16 + lr, lh);
    c00 = __builtin_amdgcn_wmma_f32_16x16x32_bf16(false, a0, false, b0, (short)0, c00, false, false);
    c01 = __builtin_amdgcn_wmma_f32_16x16x32_bf16(false, a0, false, b1, (short)0, c01, false, false);
    c10 = __builtin_amdgcn_wmma_f32_16x16x32_bf16(false, a1, false, b0, (short)0, c10, false, false);
    c11 = __builtin_amdgcn_wmma_f32_16x16x32_bf16(false, a1, false, b1, (short)0, c11, false, false);
    __syncthreads();
    buf = nxt;
  }

  // Epilogue. C layout: VGPR r -> M = r + lh*8, N = lr.
  auto emit = [&](const v8f& c, int mbase, int nbase) {
#pragma unroll
    for (int r = 0; r < 8; ++r) {
      int m  = mbase + r + lh * 8;
      int co = co0 + m;
      int w  = nbase + lr;
      float v = c[r] + Bias[co];
      if (LAYER == 0) v *= (float)((h + w) & 1);      // checkerboard (h+w)%2
      if (LAYER == 1) v = fmaxf(v, 0.0f);
      if (LAYER == 2)
        ((float*)Y)[((size_t)img * 256 + co) * 4096 + h * 64 + w] = v;
      else
        ((unsigned short*)Y)[(((size_t)img * 66 + h + 1) * 66 + (w + 1)) * 256 + co] = f2bf(v);
    }
  };
  emit(c00, mw2,      nw2);
  emit(c01, mw2,      nw2 + 16);
  emit(c10, mw2 + 16, nw2);
  emit(c11, mw2 + 16, nw2 + 16);
}

// ---------------------------- prep kernels ---------------------------------
__global__ void zero_u4_kernel(uint4* __restrict__ p, int count4) {
  const uint4 z = {0u, 0u, 0u, 0u};
  for (int i = blockIdx.x * blockDim.x + threadIdx.x; i < count4;
       i += gridDim.x * blockDim.x) p[i] = z;
}

// preds_S NCHW f32 [8][128][64][64] -> NHWC bf16 [8][4096][128]
__global__ void cvt_predsS_kernel(const float* __restrict__ in,
                                  unsigned short* __restrict__ out) {
  int idx = blockIdx.x * blockDim.x + threadIdx.x;     // 4,194,304
  int n = idx >> 19, c = (idx >> 12) & 127, hw = idx & 4095;
  out[((size_t)n * 4096 + hw) * 128 + c] = f2bf(in[idx]);
}

__global__ void cvt_walign_kernel(const float* __restrict__ in,
                                  unsigned short* __restrict__ out) {
  int idx = blockIdx.x * blockDim.x + threadIdx.x;     // 32,768
  out[idx] = f2bf(in[idx]);
}

// W_gen [co][ci][3][3] f32 -> k-major bf16 [co][tap][ci]
__global__ void cvt_wgen_kernel(const float* __restrict__ in,
                                unsigned short* __restrict__ out) {
  int idx = blockIdx.x * blockDim.x + threadIdx.x;     // 589,824 (output idx)
  int co = idx / 2304, r = idx - co * 2304;
  int tap = r >> 8, ci = r & 255;
  out[idx] = f2bf(in[co * 2304 + ci * 9 + tap]);
}

// ---------------------------- loss kernels ---------------------------------
__global__ __launch_bounds__(256)
void loss_partial_kernel(const float* __restrict__ P,   // preds_T NCHW
                         const float* __restrict__ T,   // new_fea NCHW
                         float* __restrict__ part) {
  constexpr int D = 1 << 20;
  const int bid = blockIdx.x, i = bid & 7, slab = bid >> 3;
  const int tid = threadIdx.x;
  const bool doTn = (i == 0);
  const float* pi = P + (size_t)i * D;

  float g[8] = {0,0,0,0,0,0,0,0}, tn[8] = {0,0,0,0,0,0,0,0}, pn = 0.0f;
  const int base = slab * 16384;
#pragma unroll 4
  for (int k = 0; k < 64; ++k) {
    int d = base + k * 256 + tid;
    float pv = pi[d];
    pn += pv * pv;
#pragma unroll
    for (int j = 0; j < 8; ++j) {
      float tv = T[(size_t)j * D + d];
      g[j] += pv * tv;
      if (doTn) tn[j] += tv * tv;
    }
  }
  __shared__ float red[17][256];
#pragma unroll
  for (int j = 0; j < 8; ++j) { red[j][tid] = g[j]; red[9 + j][tid] = tn[j]; }
  red[8][tid] = pn;
  __syncthreads();
  if (tid < 17) {                         // fixed order -> deterministic
    float s = 0.0f;
    for (int t = 0; t < 256; ++t) s += red[tid][t];
    part[bid * 17 + tid] = s;
  }
}

__global__ void loss_final_kernel(const float* __restrict__ part,
                                  float* __restrict__ out) {
  if (threadIdx.x != 0) return;
  float G[64], pn[8], tn[8];
  for (int k = 0; k < 64; ++k) G[k] = 0.0f;
  for (int k = 0; k < 8; ++k) { pn[k] = 0.0f; tn[k] = 0.0f; }
  for (int b = 0; b < 512; ++b) {
    int i = b & 7;
    const float* q = part + b * 17;
    for (int j = 0; j < 8; ++j) G[i * 8 + j] += q[j];
    pn[i] += q[8];
    for (int j = 0; j < 8; ++j) tn[j] += q[9 + j];
  }
  const float inv2var = 1.0f / 128.0f;
  float ce = 0.0f;
  for (int i = 0; i < 8; ++i) {
    float L[8], mx = -3.4e38f;
    for (int j = 0; j < 8; ++j) {
      L[j] = -(pn[i] + tn[j] - 2.0f * G[i * 8 + j]) * inv2var;
      mx = fmaxf(mx, L[j]);
    }
    float s = 0.0f;
    for (int j = 0; j < 8; ++j) s += __expf(L[j] - mx);
    ce += mx + __logf(s) - L[i];
  }
  out[0] = (ce / 8.0f) * 16.0f * 2e-5f;   // ce * (2*64)/8 * alpha
}

// ---------------------------------------------------------------------------
extern "C" void kernel_launch(void* const* d_in, const int* in_sizes, int n_in,
                              void* d_out, int out_size, void* d_ws, size_t ws_size,
                              hipStream_t stream)
{
  (void)in_sizes; (void)n_in; (void)out_size; (void)ws_size;
  const float* preds_S = (const float*)d_in[0];
  const float* preds_T = (const float*)d_in[1];
  const float* W_align = (const float*)d_in[2];
  const float* b_align = (const float*)d_in[3];
  const float* W_gen1  = (const float*)d_in[4];
  const float* b_gen1  = (const float*)d_in[5];
  const float* W_gen2  = (const float*)d_in[6];
  const float* b_gen2  = (const float*)d_in[7];

  char* ws = (char*)d_ws;                                   // byte offsets
  unsigned short* predsS_b = (unsigned short*)(ws);                    //  8.0 MB
  unsigned short* Wal_b    = (unsigned short*)(ws + 8388608);          // 64 KB
  unsigned short* Wg1_b    = (unsigned short*)(ws + 8454144);          // 1.125 MB
  unsigned short* Wg2_b    = (unsigned short*)(ws + 9633792);          // 1.125 MB
  unsigned short* masked_p = (unsigned short*)(ws + 10813440);         // 17.0 MB
  unsigned short* t1_p     = (unsigned short*)(ws + 28655616);         // 17.0 MB
  float*          new_fea  = (float*)        (ws + 46497792);          // 32.0 MB
  float*          part     = (float*)        (ws + 80052224);          // 34 KB

  // prep: zero both padded tensors (contiguous region), convert inputs
  zero_u4_kernel<<<4096, 256, 0, stream>>>((uint4*)masked_p, 2230272);
  cvt_predsS_kernel<<<16384, 256, 0, stream>>>(preds_S, predsS_b);
  cvt_walign_kernel<<<128, 256, 0, stream>>>(W_align, Wal_b);
  cvt_wgen_kernel<<<2304, 256, 0, stream>>>(W_gen1, Wg1_b);
  cvt_wgen_kernel<<<2304, 256, 0, stream>>>(W_gen2, Wg2_b);

  dim3 blk(256), grid(512, 2);            // 8*64 image rows, 256/128 co tiles
  conv_wmma_kernel<0><<<grid, blk, 0, stream>>>(predsS_b, Wal_b, b_align, masked_p);
  conv_wmma_kernel<1><<<grid, blk, 0, stream>>>(masked_p, Wg1_b, b_gen1, t1_p);
  conv_wmma_kernel<2><<<grid, blk, 0, stream>>>(t1_p,     Wg2_b, b_gen2, new_fea);

  loss_partial_kernel<<<512, 256, 0, stream>>>(preds_T, new_fea, part);
  loss_final_kernel<<<1, 32, 0, stream>>>(part, (float*)d_out);
}
